// ResNet___28312424415642
// MI455X (gfx1250) — matrix operand order repaired and checked by reading, not compiled, not run on hardware
//
#include <hip/hip_runtime.h>
#include <hip/hip_bf16.h>
#include <math.h>
#include <stdint.h>

// ============================================================================
// Quantized ResNet-18 forward for MI455X (gfx1250, wave32, WMMA).
//  - All convs: implicit-GEMM on v_wmma_f32_16x16x32_f16, K-step 64 (4 WMMA
//    per step), double-buffered LDS tiles.
//  - Weight (A) tile staged with GLOBAL_LOAD_ASYNC_TO_LDS_B128 (ASYNCcnt).
//  - Weights DoReFa-quantized + pattern-masked on device into fp16
//    [Cout, Kpad] packs (K = Cin*KH*KW padded to 64).
//  - Activations fp16 (4-bit quantized levels), accumulators/BN stats f32.
// ============================================================================

typedef _Float16 v8h  __attribute__((ext_vector_type(8)));
typedef _Float16 v16h __attribute__((ext_vector_type(16)));
typedef float    v8f  __attribute__((ext_vector_type(8)));
typedef int      v4i_ __attribute__((__vector_size__(16)));

#define BN_EPS 1e-5f
#define APAD 72   // LDS row pitch (f16): 144B -> 16B aligned, 36-dword stride
                  // gives distinct banks for 16 consecutive rows (conflict-free)

#define GLOBAL_AS __attribute__((address_space(1)))
#define LDS_AS    __attribute__((address_space(3)))

#if __has_builtin(__builtin_amdgcn_global_load_async_to_lds_b128)
#define HAVE_ASYNC_LDS 1
#endif

// copy 16 bytes global -> LDS (async path when available).
// AS casts done via integers: generic->global keeps the 64-bit value;
// generic->LDS keeps the LDS offset in the low 32 bits (ISA aperture rule).
__device__ __forceinline__ void copy16_g2l(const _Float16* g, _Float16* l) {
#ifdef HAVE_ASYNC_LDS
  GLOBAL_AS v4i_* gp = (GLOBAL_AS v4i_*)(unsigned long long)(uintptr_t)g;
  LDS_AS    v4i_* lp = (LDS_AS    v4i_*)(unsigned)(uintptr_t)l;
  __builtin_amdgcn_global_load_async_to_lds_b128(gp, lp, 0, 0);
#else
  *(v8h*)l = *(const v8h*)g;
#endif
}

__device__ __forceinline__ void wait_async_lds() {
#ifdef HAVE_ASYNC_LDS
#if __has_builtin(__builtin_amdgcn_s_wait_asynccnt)
  __builtin_amdgcn_s_wait_asynccnt(0);
#else
  asm volatile("s_wait_asynccnt 0x0" ::: "memory");
#endif
#endif
}

// ---------------------------------------------------------------------------
// Weight prep: max(|tanh(w)|) over tensor (single block), then pack/quantize.
// ---------------------------------------------------------------------------
__global__ __launch_bounds__(1024) void wmax_kernel(const float* __restrict__ w,
                                                    float* __restrict__ outmax, int n) {
  __shared__ float red[1024];
  float m = 0.f;
  for (int i = threadIdx.x; i < n; i += 1024) m = fmaxf(m, fabsf(tanhf(w[i])));
  red[threadIdx.x] = m;
  __syncthreads();
  for (int s = 512; s > 0; s >>= 1) {
    if (threadIdx.x < (unsigned)s) red[threadIdx.x] = fmaxf(red[threadIdx.x], red[threadIdx.x + s]);
    __syncthreads();
  }
  if (threadIdx.x == 0) outmax[0] = red[0];
}

// DoReFa: t=tanh(w); t=t/max|t|*0.5+0.5; q=round(t*15)/15; wq=2q-1; * mask
// mask==0 means "no pattern mask". Output layout [Cout, Kpad] fp16, zero pad.
__global__ __launch_bounds__(256) void wpack_kernel(const float* __restrict__ w,
                                                    const float* __restrict__ maxp,
                                                    _Float16* __restrict__ wq,
                                                    int Cout, int Kreal, int KHW,
                                                    int Kpad, unsigned mask) {
  int i = blockIdx.x * 256 + threadIdx.x;
  if (i >= Cout * Kpad) return;
  int co = i / Kpad, k = i - co * Kpad;
  float v = 0.f;
  if (k < Kreal) {
    float t = tanhf(w[co * Kreal + k]);
    t = t / maxp[0] * 0.5f + 0.5f;
    float q = rintf(t * 15.f) * (1.f / 15.f);
    v = 2.f * q - 1.f;
    if (mask) {
      int r = k % KHW;                 // kh*KW+kw position inside 3x3 / 1x1
      if (!((mask >> r) & 1u)) v = 0.f;
    }
  }
  wq[i] = (_Float16)v;
}

__global__ __launch_bounds__(256) void cast_f32_f16(const float* __restrict__ in,
                                                    _Float16* __restrict__ out, int n) {
  int i = blockIdx.x * 256 + threadIdx.x;
  if (i < n) out[i] = (_Float16)in[i];
}

// ---------------------------------------------------------------------------
// Implicit-GEMM conv with WMMA.
//   GEMM: D[M=Cout, N=B*Ho*Wo] += A[M,K] * B[K,N], K = Cin*KH*KW (Kpad mult 64)
//   Workgroup: 256 thr = 8 waves as 4(M) x 2(N); C tile 64x64; wave 16x32.
//   K-step 64: 4 v_wmma per iteration. Double-buffered LDS A / B^T tiles.
//   A tile: async global->LDS b128 (2 per thread). B tile: im2col gather.
// ---------------------------------------------------------------------------
__global__ __launch_bounds__(256) void conv_wmma_kernel(
    const _Float16* __restrict__ act,   // [B, Cin, Hi, Wi] fp16
    const _Float16* __restrict__ wq,    // [Cout, Kpad] fp16
    float* __restrict__ out,            // [B, Cout, Ho, Wo] f32
    int Cin, int Hi, int Wi, int Cout, int Ho, int Wo,
    int KH, int KW, int stride, int pad, int Kpad, int Ntot) {
  __shared__ __align__(16) _Float16 lA[2][64][APAD];
  __shared__ __align__(16) _Float16 lB[2][64][APAD];

  const int tid  = threadIdx.x;
  const int lane = tid & 31;
  const int wave = tid >> 5;      // 0..7
  const int wm   = wave >> 1;     // 0..3 : M sub-tile
  const int wn   = wave & 1;      // 0..1 : N sub-tile
  const int tileM = blockIdx.y * 64;
  const int tileN = blockIdx.x * 64;
  const int Kreal = Cin * KH * KW;
  const int KHW   = KH * KW;
  const int HW    = Ho * Wo;

  // staging assignment: thread owns 16 f16 of one row (two 16B chunks)
  const int srow = tid >> 2;          // 0..63
  const int scol = (tid & 3) * 16;    // 0,16,32,48

  // im2col coordinates for this thread's N row (fixed across K loop)
  const int ng = tileN + srow;
  const bool nok = ng < Ntot;
  int bb = 0, ho = 0, wo = 0;
  if (nok) { bb = ng / HW; int r = ng - bb * HW; ho = r / Wo; wo = r - ho * Wo; }
  const int ibase = ho * stride - pad;
  const int jbase = wo * stride - pad;

  v8f c0 = {}; v8f c1 = {};
  const int nK = Kpad >> 6;

  // stage k-block `ks` into LDS buffer `dst`
  auto stage = [&](int ks, int dst) {
    const int k0 = ks * 64;
    // ---- A: weights, regular addresses -> async LDS copy ----
    const _Float16* ga = wq + (size_t)(tileM + srow) * Kpad + k0 + scol;
    copy16_g2l(ga,     &lA[dst][srow][scol]);
    copy16_g2l(ga + 8, &lA[dst][srow][scol + 8]);
    // ---- B: im2col gather with zero fill ----
    v8h b0r = {}, b1r = {};
#pragma unroll
    for (int j = 0; j < 16; j++) {
      int k = k0 + scol + j;
      _Float16 v = (_Float16)0.f;
      if (nok && k < Kreal) {
        int cin = k / KHW; int r = k - cin * KHW; int kh = r / KW; int kw = r - kh * KW;
        int ii = ibase + kh, jj = jbase + kw;
        if (ii >= 0 && ii < Hi && jj >= 0 && jj < Wi)
          v = act[((size_t)(bb * Cin + cin) * Hi + ii) * Wi + jj];
      }
      if (j < 8) b0r[j] = v; else b1r[j - 8] = v;
    }
    *(v8h*)&lB[dst][srow][scol]     = b0r;
    *(v8h*)&lB[dst][srow][scol + 8] = b1r;
  };

  // ---- prologue ----
  stage(0, 0);
  wait_async_lds();
  __syncthreads();

  const int h  = lane >> 4;   // half-wave id (K offset selector)
  const int ml = lane & 15;   // M (for A) / N (for B) local index

  for (int ks = 0; ks < nK; ks++) {
    const int buf = ks & 1;
    if (ks + 1 < nK) stage(ks + 1, buf ^ 1);   // overlaps with WMMA below

#pragma unroll
    for (int t = 0; t < 2; t++) {
      const int kt = t * 32;
      // A fragment (16-bit A 16x32): lane m=ml, half h:
      //   elem e -> K = kt+e+16h
      v8h alo = *(const v8h*)&lA[buf][wm * 16 + ml][kt + 16 * h];
      v8h ahi = *(const v8h*)&lA[buf][wm * 16 + ml][kt + 16 * h + 8];
      v16h a;
#pragma unroll
      for (int i = 0; i < 8; i++) { a[i] = alo[i]; a[8 + i] = ahi[i]; }

      // B fragments (16-bit B 32x16): lane n=ml, half h: elem e -> K = kt+e+16h
      v8h b0lo = *(const v8h*)&lB[buf][wn * 32 + ml][kt + 16 * h];
      v8h b0hi = *(const v8h*)&lB[buf][wn * 32 + ml][kt + 16 * h + 8];
      v8h b1lo = *(const v8h*)&lB[buf][wn * 32 + 16 + ml][kt + 16 * h];
      v8h b1hi = *(const v8h*)&lB[buf][wn * 32 + 16 + ml][kt + 16 * h + 8];
      v16h b0v, b1v;
#pragma unroll
      for (int i = 0; i < 8; i++) {
        b0v[i] = b0lo[i]; b0v[8 + i] = b0hi[i];
        b1v[i] = b1lo[i]; b1v[8 + i] = b1hi[i];
      }

      c0 = __builtin_amdgcn_wmma_f32_16x16x32_f16(false, a, false, b0v, (short)0, c0, false, false);
      c1 = __builtin_amdgcn_wmma_f32_16x16x32_f16(false, a, false, b1v, (short)0, c1, false, false);
    }

    wait_async_lds();   // A-tile async copies for next buffer done
    __syncthreads();
  }

  // ---- writeback: C layout: lane l: N=l&15, M=(l>>4)*8 + vgpr ----
  int n0 = tileN + wn * 32 + ml;
  int n1 = n0 + 16;
  int b0n = 0, hw0 = 0, b1n = 0, hw1 = 0;
  bool v0 = n0 < Ntot, v1 = n1 < Ntot;
  if (v0) { b0n = n0 / HW; hw0 = n0 - b0n * HW; }
  if (v1) { b1n = n1 / HW; hw1 = n1 - b1n * HW; }
#pragma unroll
  for (int r = 0; r < 8; r++) {
    int m = tileM + wm * 16 + h * 8 + r;
    if (v0) out[((size_t)b0n * Cout + m) * HW + hw0] = c0[r];
    if (v1) out[((size_t)b1n * Cout + m) * HW + hw1] = c1[r];
  }
}

// ---------------------------------------------------------------------------
// BatchNorm batch statistics: per-channel mean/var over (B, H, W).
// ---------------------------------------------------------------------------
__global__ __launch_bounds__(256) void bnstats_kernel(const float* __restrict__ x,
                                                      float* __restrict__ mean,
                                                      float* __restrict__ var,
                                                      int C, int Bn, int HW) {
  __shared__ float s1[256], s2[256];
  const int c = blockIdx.x;
  float a = 0.f, b = 0.f;
  for (int bb = 0; bb < Bn; bb++) {
    const float* p = x + ((size_t)bb * C + c) * HW;
    for (int hw = threadIdx.x; hw < HW; hw += 256) {
      float v = p[hw];
      a += v; b += v * v;
    }
  }
  s1[threadIdx.x] = a; s2[threadIdx.x] = b;
  __syncthreads();
  for (int s = 128; s > 0; s >>= 1) {
    if (threadIdx.x < (unsigned)s) {
      s1[threadIdx.x] += s1[threadIdx.x + s];
      s2[threadIdx.x] += s2[threadIdx.x + s];
    }
    __syncthreads();
  }
  if (threadIdx.x == 0) {
    const int cnt = Bn * HW;
    float m = s1[0] / cnt;
    mean[c] = m;
    var[c] = s2[0] / cnt - m * m;
  }
}

// mode 0: relu ; mode 1: quant_act (clip[0,1], 4-bit round-to-even)
__global__ __launch_bounds__(256) void bn_apply_kernel(const float* __restrict__ x,
                                                       const float* __restrict__ mean,
                                                       const float* __restrict__ var,
                                                       const float* __restrict__ g,
                                                       const float* __restrict__ bta,
                                                       _Float16* __restrict__ y,
                                                       int C, int HW, int total, int mode) {
  int i = blockIdx.x * 256 + threadIdx.x;
  if (i >= total) return;
  int c = (i / HW) % C;
  float z = g[c] * (x[i] - mean[c]) * rsqrtf(var[c] + BN_EPS) + bta[c];
  if (mode == 0) {
    z = fmaxf(z, 0.f);
  } else {
    z = fminf(fmaxf(z, 0.f), 1.f);
    z = rintf(z * 15.f) * (1.f / 15.f);
  }
  y[i] = (_Float16)z;
}

// out = quant_act( BN(h) + identity ); identity = fp16 tensor or BN(ds conv)
__global__ __launch_bounds__(256) void bn_residual_kernel(const float* h,
                                                          const float* mh, const float* vh,
                                                          const float* g, const float* bta,
                                                          const _Float16* idh,
                                                          const float* ds,
                                                          const float* md, const float* vd,
                                                          const float* dg, const float* db,
                                                          int hasds, _Float16* y,
                                                          int C, int HW, int total) {
  int i = blockIdx.x * 256 + threadIdx.x;
  if (i >= total) return;
  int c = (i / HW) % C;
  float z = g[c] * (h[i] - mh[c]) * rsqrtf(vh[c] + BN_EPS) + bta[c];
  float idv;
  if (hasds) idv = dg[c] * (ds[i] - md[c]) * rsqrtf(vd[c] + BN_EPS) + db[c];
  else       idv = (float)idh[i];
  z += idv;
  z = fminf(fmaxf(z, 0.f), 1.f);
  z = rintf(z * 15.f) * (1.f / 15.f);
  y[i] = (_Float16)z;
}

__global__ __launch_bounds__(256) void maxpool_kernel(const _Float16* __restrict__ in,
                                                      _Float16* __restrict__ out,
                                                      int C, int Hi, int Wi, int Ho, int Wo,
                                                      int total) {
  int i = blockIdx.x * 256 + threadIdx.x;
  if (i >= total) return;
  int wo = i % Wo; int t = i / Wo; int ho = t % Ho; t /= Ho; int c = t % C; int b = t / C;
  float m = -1e30f;
#pragma unroll
  for (int kh = 0; kh < 3; kh++) {
#pragma unroll
    for (int kw = 0; kw < 3; kw++) {
      int ii = ho * 2 - 1 + kh, jj = wo * 2 - 1 + kw;
      if (ii >= 0 && ii < Hi && jj >= 0 && jj < Wi)
        m = fmaxf(m, (float)in[((size_t)(b * C + c) * Hi + ii) * Wi + jj]);
    }
  }
  out[i] = (_Float16)m;
}

__global__ __launch_bounds__(256) void avgpool_kernel(const _Float16* __restrict__ in,
                                                      float* __restrict__ out,
                                                      int HW, int total) {
  int i = blockIdx.x * 256 + threadIdx.x;   // i = b*C + c
  if (i >= total) return;
  float s = 0.f;
  for (int j = 0; j < HW; j++) s += (float)in[(size_t)i * HW + j];
  out[i] = s / (float)HW;
}

__global__ __launch_bounds__(256) void fc_kernel(const float* __restrict__ x,
                                                 const float* __restrict__ w,
                                                 const float* __restrict__ bias,
                                                 float* __restrict__ out,
                                                 int K, int O, int total) {
  int i = blockIdx.x * 256 + threadIdx.x;   // i = b*O + o
  if (i >= total) return;
  int o = i % O, b = i / O;
  float s = bias[o];
  for (int k = 0; k < K; k++) s += x[b * K + k] * w[o * K + k];
  out[i] = s;
}

// ---------------------------------------------------------------------------
// Host orchestration
// ---------------------------------------------------------------------------
static void launch_conv(const _Float16* act, const _Float16* wq, float* out,
                        int Bn, int Cin, int Hi, int Wi, int Cout, int Ho, int Wo,
                        int KH, int KW, int stride, int pad, int Kpad, hipStream_t s) {
  int Ntot = Bn * Ho * Wo;
  dim3 grid((Ntot + 63) / 64, Cout / 64);
  conv_wmma_kernel<<<grid, 256, 0, s>>>(act, wq, out, Cin, Hi, Wi, Cout, Ho, Wo,
                                        KH, KW, stride, pad, Kpad, Ntot);
}

extern "C" void kernel_launch(void* const* d_in, const int* in_sizes, int n_in,
                              void* d_out, int out_size, void* d_ws, size_t ws_size,
                              hipStream_t stream) {
  (void)in_sizes; (void)n_in; (void)out_size; (void)ws_size;
  const int Bn = 32;

  // ---- input index map (setup_inputs insertion order, depth-first) ----
  const int IDX_X = 0, IDX_STEM_W = 1, IDX_STEM_G = 2, IDX_STEM_B = 3;
  const int blockBase[8] = {4, 10, 16, 25, 31, 40, 46, 55};
  const int IDX_FC_W = 61, IDX_FC_B = 62;

  // PatDNN 3x3 pattern bitmasks (bit = kh*3+kw)
  const unsigned pat1[4] = {0x0BAu, 0x0B6u, 0x1B4u, 0x1FFu};
  const unsigned pat2[4] = {0x0B3u, 0x19Au, 0x11Bu, 0x1FFu};
  const int planes_[4] = {64, 128, 256, 512};

  // ---- workspace arena ----
  char* base = (char*)d_ws;
  size_t off = 0;
  auto alloc = [&](size_t bytes) -> void* {
    void* p = base + off;
    off += bytes;
    off = (off + 255) & ~(size_t)255;
    return p;
  };
  auto kpad64 = [](int k) { return (k + 63) & ~63; };

  float* maxscr = (float*)alloc(16 * sizeof(float));

  // device-side weight prep (sequential on stream -> maxscr reuse is safe)
  auto prep = [&](int idx, int Cout, int Cin, int KH, int KW, unsigned mask) -> _Float16* {
    int Kreal = Cin * KH * KW;
    int Kpad = kpad64(Kreal);
    _Float16* wq = (_Float16*)alloc((size_t)Cout * Kpad * sizeof(_Float16));
    wmax_kernel<<<1, 1024, 0, stream>>>((const float*)d_in[idx], maxscr, Cout * Kreal);
    int tot = Cout * Kpad;
    wpack_kernel<<<(tot + 255) / 256, 256, 0, stream>>>((const float*)d_in[idx], maxscr, wq,
                                                        Cout, Kreal, KH * KW, Kpad, mask);
    return wq;
  };

  _Float16* wq_stem = prep(IDX_STEM_W, 64, 3, 7, 7, 0);
  _Float16* wq1[8]; _Float16* wq2[8]; _Float16* wqd[4] = {nullptr, nullptr, nullptr, nullptr};
  {
    int cin = 64;
    for (int si = 0; si < 4; si++) {
      int p = planes_[si];
      for (int bi = 0; bi < 2; bi++) {
        int bidx = blockBase[si * 2 + bi];
        int ci = (bi == 0) ? cin : p;
        bool first = (bi == 0 && ci != p);
        unsigned m1 = first ? 0u : pat1[si];
        unsigned m2 = first ? pat1[si] : pat2[si];
        wq1[si * 2 + bi] = prep(bidx + 0, p, ci, 3, 3, m1);
        wq2[si * 2 + bi] = prep(bidx + 3, p, p, 3, 3, m2);
        if (first) wqd[si] = prep(bidx + 6, p, ci, 1, 1, 0);
      }
      cin = p;
    }
  }

  float* mean0 = (float*)alloc(512 * sizeof(float));
  float* var0  = (float*)alloc(512 * sizeof(float));
  float* mean1 = (float*)alloc(512 * sizeof(float));
  float* var1  = (float*)alloc(512 * sizeof(float));
  _Float16* act0 = (_Float16*)alloc((size_t)Bn * 64 * 56 * 56 * sizeof(_Float16));   // 6.4M elems (also holds x fp16: 4.8M)
  _Float16* act1 = (_Float16*)alloc((size_t)Bn * 64 * 112 * 112 * sizeof(_Float16)); // 25.7M elems
  float* convf = (float*)alloc((size_t)Bn * 64 * 112 * 112 * sizeof(float));
  float* dsf   = (float*)alloc((size_t)Bn * 128 * 28 * 28 * sizeof(float));
  float* pooled = (float*)alloc((size_t)Bn * 512 * sizeof(float));

  // ---- stem: conv7x7/2 -> BN(relu) -> maxpool3x3/2 ----
  {
    int n = Bn * 3 * 224 * 224;
    cast_f32_f16<<<(n + 255) / 256, 256, 0, stream>>>((const float*)d_in[IDX_X], act0, n);
  }
  launch_conv(act0, wq_stem, convf, Bn, 3, 224, 224, 64, 112, 112, 7, 7, 2, 3, kpad64(3 * 49), stream);
  bnstats_kernel<<<64, 256, 0, stream>>>(convf, mean0, var0, 64, Bn, 112 * 112);
  {
    int tot = Bn * 64 * 112 * 112;
    bn_apply_kernel<<<(tot + 255) / 256, 256, 0, stream>>>(convf, mean0, var0,
        (const float*)d_in[IDX_STEM_G], (const float*)d_in[IDX_STEM_B],
        act1, 64, 112 * 112, tot, 0);
  }
  {
    int tot = Bn * 64 * 56 * 56;
    maxpool_kernel<<<(tot + 255) / 256, 256, 0, stream>>>(act1, act0, 64, 112, 112, 56, 56, tot);
  }

  // ---- residual stages ----
  int cin = 64, H = 56;
  for (int si = 0; si < 4; si++) {
    int p = planes_[si];
    for (int bi = 0; bi < 2; bi++) {
      int bidx = blockBase[si * 2 + bi];
      int stride = (si > 0 && bi == 0) ? 2 : 1;
      int ci = (bi == 0) ? cin : p;
      bool first = (bi == 0 && ci != p);
      int Ho = H / stride;
      int HW = Ho * Ho;
      int tot = Bn * p * HW;

      // conv1 + BN + quant_act -> act1
      launch_conv(act0, wq1[si * 2 + bi], convf, Bn, ci, H, H, p, Ho, Ho, 3, 3, stride, 1,
                  kpad64(ci * 9), stream);
      bnstats_kernel<<<p, 256, 0, stream>>>(convf, mean0, var0, p, Bn, HW);
      bn_apply_kernel<<<(tot + 255) / 256, 256, 0, stream>>>(convf, mean0, var0,
          (const float*)d_in[bidx + 1], (const float*)d_in[bidx + 2], act1, p, HW, tot, 1);

      // conv2 -> convf ; its BN stats
      launch_conv(act1, wq2[si * 2 + bi], convf, Bn, p, Ho, Ho, p, Ho, Ho, 3, 3, 1, 1,
                  kpad64(p * 9), stream);
      bnstats_kernel<<<p, 256, 0, stream>>>(convf, mean0, var0, p, Bn, HW);

      if (first) {
        // downsample 1x1/stride conv + BN as identity
        launch_conv(act0, wqd[si], dsf, Bn, ci, H, H, p, Ho, Ho, 1, 1, stride, 0,
                    kpad64(ci), stream);
        bnstats_kernel<<<p, 256, 0, stream>>>(dsf, mean1, var1, p, Bn, HW);
        bn_residual_kernel<<<(tot + 255) / 256, 256, 0, stream>>>(convf, mean0, var0,
            (const float*)d_in[bidx + 4], (const float*)d_in[bidx + 5],
            nullptr, dsf, mean1, var1,
            (const float*)d_in[bidx + 7], (const float*)d_in[bidx + 8],
            1, act0, p, HW, tot);
      } else {
        bn_residual_kernel<<<(tot + 255) / 256, 256, 0, stream>>>(convf, mean0, var0,
            (const float*)d_in[bidx + 4], (const float*)d_in[bidx + 5],
            act0, nullptr, nullptr, nullptr, nullptr, nullptr,
            0, act0, p, HW, tot);
      }
      H = Ho;
    }
    cin = p;
  }

  // ---- avgpool + fc ----
  {
    int tot = Bn * 512;
    avgpool_kernel<<<(tot + 255) / 256, 256, 0, stream>>>(act0, pooled, 49, tot);
  }
  {
    int tot = Bn * 1000;
    fc_kernel<<<(tot + 255) / 256, 256, 0, stream>>>(pooled, (const float*)d_in[IDX_FC_W],
        (const float*)d_in[IDX_FC_B], (float*)d_out, 512, 1000, tot);
  }
}
